// DeepSeekMegaBlocksAdapter_82617990906328
// MI455X (gfx1250) — matrix operand order repaired and checked by reading, not compiled, not run on hardware
//
#include <hip/hip_runtime.h>
#include <hip/hip_bf16.h>

// Problem constants (reference shapes)
#define T_NUM 2048   // B*S tokens
#define H_DIM 2048
#define F_DIM 1024
#define FS_DIM 2048
#define E_NUM 16
#define TOPK 4

typedef __attribute__((ext_vector_type(16))) __bf16 v16bf;
typedef __attribute__((ext_vector_type(8)))  float  v8f;
typedef __attribute__((ext_vector_type(2)))  __bf16 bf16x2;
typedef __attribute__((ext_vector_type(4)))  __bf16 bf16x4;

__device__ __forceinline__ float silu_f(float v) {
    // fast: x * rcp(1+exp(-x)) -> v_exp_f32 + v_rcp_f32 instead of IEEE div chain
    return v * __builtin_amdgcn_rcpf(1.0f + __expf(-v));
}

__device__ __forceinline__ bf16x4 cvt4(float4 f) {
    bf16x4 r;
    r.x = (__bf16)f.x; r.y = (__bf16)f.y; r.z = (__bf16)f.z; r.w = (__bf16)f.w;
    return r;
}

// ---- WMMA fragment loaders (ISA 7.12.2 layouts, wave32) -------------------
// A tile: 16x32 bf16 row-major in LDS (row stride 32). Lane L holds row
// M=L%16; its 16 elements are K = 8*half+{0..7} then 16+8*half+{0..7}:
// two contiguous 16-byte runs -> 2x ds_load_b128.
__device__ __forceinline__ v16bf load_a_frag(const __bf16* As, int lane) {
    int m = lane & 15, half = lane >> 4;
    union { uint4 q[2]; v16bf v; } u;
    u.q[0] = *(const uint4*)(As + m * 32 + 8 * half);
    u.q[1] = *(const uint4*)(As + m * 32 + 16 + 8 * half);
    return u.v;
}

// B tile stored TRANSPOSED in LDS: Bt[n][k], 64 rows x 32 cols (row stride 32).
// Lane L holds column N=L%16 (+n0); elements are K = 8*half+{0..7} then 16+8*half+{0..7}:
// two contiguous 16-byte runs -> 2x ds_load_b128.
__device__ __forceinline__ v16bf load_bt_frag(const __bf16* Bt, int n0, int lane) {
    int n = (lane & 15) + n0, half = lane >> 4;
    union { uint4 q[2]; v16bf v; } u;
    const __bf16* p = Bt + n * 32 + 8 * half;
    u.q[0] = *(const uint4*)(p);
    u.q[1] = *(const uint4*)(p + 16);
    return u.v;
}

// ---- Tensor Data Mover: async 2D bf16 tile -> LDS (gfx1250 TDM) -----------
#if defined(__gfx1250__) && __has_builtin(__builtin_amdgcn_tensor_load_to_lds)
#define HAVE_TDM 1
typedef __attribute__((ext_vector_type(4))) unsigned int u32x4;
typedef __attribute__((ext_vector_type(8))) int i32x8;
typedef __attribute__((ext_vector_type(4))) int i32x4;

__device__ __forceinline__ void tdm_load_2d_bf16(const __bf16* gsrc, __bf16* ldst,
                                                 int tile_cols, int tile_rows,
                                                 int row_stride_elems) {
    unsigned long long ga = (unsigned long long)(size_t)gsrc;
    unsigned int lds = (unsigned int)(size_t)ldst;   // LDS addr = low 32 bits
    // D# group 0: count=1 | lds_addr | global_addr[56:0] | type=2
    u32x4 g0;
    g0[0] = 1u;
    g0[1] = lds;
    g0[2] = (unsigned int)ga;
    g0[3] = (unsigned int)((ga >> 32) & 0x01FFFFFFull) | (2u << 30);
    // D# group 1: data_size=1(2B); tensor_dim0=stride; tensor_dim1=rows;
    // tile_dim0=cols; tile_dim1=rows; tensor_dim0_stride=stride
    unsigned td0 = (unsigned)row_stride_elems;
    unsigned td1 = (unsigned)tile_rows;
    i32x8 g1;
    g1[0] = (int)(1u << 16);
    g1[1] = (int)((td0 & 0xFFFFu) << 16);
    g1[2] = (int)(((td0 >> 16) & 0xFFFFu) | ((td1 & 0xFFFFu) << 16));
    g1[3] = (int)(((td1 >> 16) & 0xFFFFu) | (((unsigned)tile_cols & 0xFFFFu) << 16));
    g1[4] = (int)((unsigned)tile_rows & 0xFFFFu);
    g1[5] = (int)td0;
    g1[6] = 0;
    g1[7] = 0;
    i32x4 gz = {0, 0, 0, 0};
#if __clang_major__ >= 23
    i32x8 gz8 = {0, 0, 0, 0, 0, 0, 0, 0};
    __builtin_amdgcn_tensor_load_to_lds(g0, g1, gz, gz, gz8, 0);
#else
    __builtin_amdgcn_tensor_load_to_lds(g0, g1, gz, gz, 0);
#endif
    __builtin_amdgcn_s_wait_tensorcnt(0);   // TDM write to LDS complete
}
#endif

// ---------------------------------------------------------------------------
// 1) Router: logits -> softmax -> top4 -> L1 renorm -> per-expert lists
// ---------------------------------------------------------------------------
__global__ __launch_bounds__(256) void k_router(const float* __restrict__ x,
                                                const float* __restrict__ rw,
                                                int* __restrict__ counts,
                                                int* __restrict__ tok_list,
                                                float* __restrict__ gate_list) {
    __shared__ float sc[16][17];
    int tid = threadIdx.x;
    int lt = tid >> 4, e = tid & 15;
    int t = blockIdx.x * 16 + lt;

    const float4* xr = (const float4*)(x + (size_t)t * H_DIM);
    const float4* wr = (const float4*)(rw + (size_t)e * H_DIM);
    float s = 0.f;
#pragma unroll 4
    for (int i = 0; i < H_DIM / 4; ++i) {
        float4 a = xr[i], b = wr[i];
        s += a.x * b.x + a.y * b.y + a.z * b.z + a.w * b.w;
    }
    sc[lt][e] = s;
    __syncthreads();

    if (e == 0) {
        float mx = -1e30f;
#pragma unroll
        for (int j = 0; j < E_NUM; ++j) mx = fmaxf(mx, sc[lt][j]);
        float p[E_NUM];
        float sum = 0.f;
#pragma unroll
        for (int j = 0; j < E_NUM; ++j) { p[j] = __expf(sc[lt][j] - mx); sum += p[j]; }
        float inv = __builtin_amdgcn_rcpf(sum);
#pragma unroll
        for (int j = 0; j < E_NUM; ++j) p[j] *= inv;

        int   idx[TOPK];
        float w[TOPK];
        float wsum = 0.f;
#pragma unroll
        for (int kk = 0; kk < TOPK; ++kk) {
            int bi = 0; float bv = -1.f;
#pragma unroll
            for (int j = 0; j < E_NUM; ++j)
                if (p[j] > bv) { bv = p[j]; bi = j; }
            idx[kk] = bi; w[kk] = bv; wsum += bv; p[bi] = -2.f;
        }
        float winv = 1.f / wsum;   // keep precise: feeds every output element
#pragma unroll
        for (int kk = 0; kk < TOPK; ++kk) {
            int pos = atomicAdd(&counts[idx[kk]], 1);
            tok_list[idx[kk] * T_NUM + pos]  = t;
            gate_list[idx[kk] * T_NUM + pos] = w[kk] * winv;
        }
    }
}

// ---------------------------------------------------------------------------
// 2) Shared expert GEMM1: act_s = silu(x@sg^T) * (x@su^T), bf16 out
// grid (FS/64, T/16), block 128 (4 waves, each one 16x16 N-subtile)
// ---------------------------------------------------------------------------
__global__ __launch_bounds__(128) void k_shared_gemm1(const float* __restrict__ x,
                                                      const float* __restrict__ sg,
                                                      const float* __restrict__ su,
                                                      __bf16* __restrict__ acts) {
    int mt = blockIdx.y, n0 = blockIdx.x * 64;
    int tid = threadIdx.x, lane = tid & 31, wave = tid >> 5;

    __shared__ __align__(16) __bf16 As[16 * 32];
    __shared__ __align__(16) __bf16 B1t[64 * 32];   // transposed [n][k]
    __shared__ __align__(16) __bf16 B2t[64 * 32];

    v8f acc1 = {0, 0, 0, 0, 0, 0, 0, 0};
    v8f acc2 = {0, 0, 0, 0, 0, 0, 0, 0};

    int arow = tid >> 3, acol = (tid & 7) * 4;
    const float* xrow = x + (size_t)(mt * 16 + arow) * H_DIM + acol;

    int bcol = tid & 63, ks = (tid >> 6) * 16;      // 16 contiguous K per thread
    const float* sgrow = sg + (size_t)(n0 + bcol) * H_DIM + ks;
    const float* surow = su + (size_t)(n0 + bcol) * H_DIM + ks;

#pragma unroll 2
    for (int k0 = 0; k0 < H_DIM; k0 += 32) {
        *(bf16x4*)&As[arow * 32 + acol] = cvt4(*(const float4*)(xrow + k0));
#pragma unroll
        for (int i = 0; i < 4; ++i) {   // sources contiguous along K -> packed stores
            *(bf16x4*)&B1t[bcol * 32 + ks + i * 4] = cvt4(*(const float4*)(sgrow + k0 + i * 4));
            *(bf16x4*)&B2t[bcol * 32 + ks + i * 4] = cvt4(*(const float4*)(surow + k0 + i * 4));
        }
        __syncthreads();
        v16bf a  = load_a_frag(As, lane);
        v16bf b1 = load_bt_frag(B1t, wave * 16, lane);
        v16bf b2 = load_bt_frag(B2t, wave * 16, lane);
        acc1 = __builtin_amdgcn_wmma_f32_16x16x32_bf16(false, a, false, b1, (short)0, acc1, false, false);
        acc2 = __builtin_amdgcn_wmma_f32_16x16x32_bf16(false, a, false, b2, (short)0, acc2, false, false);
        __syncthreads();
    }

    int half = lane >> 4, n = lane & 15;
#pragma unroll
    for (int j = 0; j < 8; ++j) {
        int m = j + 8 * half;
        float h = silu_f(acc1[j]) * acc2[j];
        acts[(size_t)(mt * 16 + m) * FS_DIM + n0 + wave * 16 + n] = (__bf16)h;
    }
}

// ---------------------------------------------------------------------------
// 3) Shared expert GEMM2: out = act_s @ sd^T (plain store, initializes out)
// grid (H/64, T/16), block 128
// ---------------------------------------------------------------------------
__global__ __launch_bounds__(128) void k_shared_gemm2(const __bf16* __restrict__ acts,
                                                      const float* __restrict__ sd,
                                                      float* __restrict__ out) {
    int mt = blockIdx.y, n0 = blockIdx.x * 64;
    int tid = threadIdx.x, lane = tid & 31, wave = tid >> 5;

    __shared__ __align__(16) __bf16 As[16 * 32];
    __shared__ __align__(16) __bf16 Bt[64 * 32];

    v8f acc = {0, 0, 0, 0, 0, 0, 0, 0};

    int arow = tid >> 3, acol = (tid & 7) * 4;
    const __bf16* atile0 = acts + (size_t)(mt * 16) * FS_DIM;         // tile origin (TDM)
    const __bf16* actrow = atile0 + (size_t)arow * FS_DIM + acol;     // fallback path

    int bcol = tid & 63, ks = (tid >> 6) * 16;
    const float* sdrow = sd + (size_t)(n0 + bcol) * FS_DIM + ks;

#pragma unroll 2
    for (int k0 = 0; k0 < FS_DIM; k0 += 32) {
#ifdef HAVE_TDM
        if (tid == 0)   // one TDM descriptor moves the whole 16x32 bf16 A tile
            tdm_load_2d_bf16(atile0 + k0, As, 32, 16, FS_DIM);
        (void)actrow;
#else
        *(uint2*)&As[arow * 32 + acol] = *(const uint2*)(actrow + k0);
#endif
#pragma unroll
        for (int i = 0; i < 4; ++i)
            *(bf16x4*)&Bt[bcol * 32 + ks + i * 4] = cvt4(*(const float4*)(sdrow + k0 + i * 4));
        __syncthreads();
        v16bf a = load_a_frag(As, lane);
        v16bf b = load_bt_frag(Bt, wave * 16, lane);
        acc = __builtin_amdgcn_wmma_f32_16x16x32_bf16(false, a, false, b, (short)0, acc, false, false);
        __syncthreads();
    }

    int half = lane >> 4, n = lane & 15;
#pragma unroll
    for (int j = 0; j < 8; ++j) {
        int m = j + 8 * half;
        out[(size_t)(mt * 16 + m) * H_DIM + n0 + wave * 16 + n] = acc[j];
    }
}

// ---------------------------------------------------------------------------
// 4) Routed GEMM1 (gathered): h = silu(x_g@w1[e]) * (x_g@v1[e]), bf16 out
// grid (F/64, T/16, E); blocks past expert's row count exit immediately.
// ---------------------------------------------------------------------------
__global__ __launch_bounds__(128) void k_moe_gemm1(const float* __restrict__ x,
                                                   const float* __restrict__ w1,
                                                   const float* __restrict__ v1,
                                                   const int* __restrict__ counts,
                                                   const int* __restrict__ tok_list,
                                                   __bf16* __restrict__ act) {
    int e = blockIdx.z;
    int cnt = counts[e];
    int mt = blockIdx.y;
    if (mt * 16 >= cnt) return;
    int n0 = blockIdx.x * 64;
    int tid = threadIdx.x, lane = tid & 31, wave = tid >> 5;

    __shared__ __align__(16) __bf16 As[16 * 32];
    __shared__ __align__(16) __bf16 B1t[64 * 32];   // transposed [n][k]
    __shared__ __align__(16) __bf16 B2t[64 * 32];
    __shared__ int toks[16];

    if (tid < 16) {
        int r = mt * 16 + tid;
        toks[tid] = tok_list[e * T_NUM + min(r, cnt - 1)];  // clamp padding rows
    }
    __syncthreads();

    v8f acc1 = {0, 0, 0, 0, 0, 0, 0, 0};
    v8f acc2 = {0, 0, 0, 0, 0, 0, 0, 0};

    int arow = tid >> 3, acol = (tid & 7) * 4;
    const float* xrow = x + (size_t)toks[arow] * H_DIM + acol;
    const float* w1base = w1 + (size_t)e * H_DIM * F_DIM + n0;
    const float* v1base = v1 + (size_t)e * H_DIM * F_DIM + n0;

#pragma unroll 2
    for (int k0 = 0; k0 < H_DIM; k0 += 32) {
        *(bf16x4*)&As[arow * 32 + acol] = cvt4(*(const float4*)(xrow + k0));
        // B tiles row-major in global: thread owns a row-pair x 4 columns so the
        // transposed LDS stores are packed K-pairs (ds_store_b32).
#pragma unroll
        for (int i = 0; i < 2; ++i) {
            int unit = tid + i * 128;          // 0..255 units
            int rp = unit >> 4;                // rows 2rp, 2rp+1
            int c4 = (unit & 15) * 4;
            const float* r0 = w1base + (size_t)(k0 + 2 * rp) * F_DIM + c4;
            float4 a0 = *(const float4*)r0;
            float4 a1 = *(const float4*)(r0 + F_DIM);
            *(bf16x2*)&B1t[(c4 + 0) * 32 + 2 * rp] = bf16x2{(__bf16)a0.x, (__bf16)a1.x};
            *(bf16x2*)&B1t[(c4 + 1) * 32 + 2 * rp] = bf16x2{(__bf16)a0.y, (__bf16)a1.y};
            *(bf16x2*)&B1t[(c4 + 2) * 32 + 2 * rp] = bf16x2{(__bf16)a0.z, (__bf16)a1.z};
            *(bf16x2*)&B1t[(c4 + 3) * 32 + 2 * rp] = bf16x2{(__bf16)a0.w, (__bf16)a1.w};
            const float* s0 = v1base + (size_t)(k0 + 2 * rp) * F_DIM + c4;
            float4 b0 = *(const float4*)s0;
            float4 b1v = *(const float4*)(s0 + F_DIM);
            *(bf16x2*)&B2t[(c4 + 0) * 32 + 2 * rp] = bf16x2{(__bf16)b0.x, (__bf16)b1v.x};
            *(bf16x2*)&B2t[(c4 + 1) * 32 + 2 * rp] = bf16x2{(__bf16)b0.y, (__bf16)b1v.y};
            *(bf16x2*)&B2t[(c4 + 2) * 32 + 2 * rp] = bf16x2{(__bf16)b0.z, (__bf16)b1v.z};
            *(bf16x2*)&B2t[(c4 + 3) * 32 + 2 * rp] = bf16x2{(__bf16)b0.w, (__bf16)b1v.w};
        }
        if (k0 + 32 < H_DIM)                   // global_prefetch_b8 next tile
            __builtin_prefetch(w1base + (size_t)(k0 + 32) * F_DIM, 0, 1);
        __syncthreads();
        v16bf a  = load_a_frag(As, lane);
        v16bf b1 = load_bt_frag(B1t, wave * 16, lane);
        v16bf b2 = load_bt_frag(B2t, wave * 16, lane);
        acc1 = __builtin_amdgcn_wmma_f32_16x16x32_bf16(false, a, false, b1, (short)0, acc1, false, false);
        acc2 = __builtin_amdgcn_wmma_f32_16x16x32_bf16(false, a, false, b2, (short)0, acc2, false, false);
        __syncthreads();
    }

    int half = lane >> 4, n = lane & 15;
#pragma unroll
    for (int j = 0; j < 8; ++j) {
        int m = j + 8 * half;
        float h = silu_f(acc1[j]) * acc2[j];
        act[((size_t)e * T_NUM + mt * 16 + m) * F_DIM + n0 + wave * 16 + n] = (__bf16)h;
    }
}

// ---------------------------------------------------------------------------
// 5) Routed GEMM2: out[tok,:] += gate * (h @ w2[e]) via f32 atomics
// grid (H/64, T/16, E)
// ---------------------------------------------------------------------------
__global__ __launch_bounds__(128) void k_moe_gemm2(const __bf16* __restrict__ act,
                                                   const float* __restrict__ w2,
                                                   const int* __restrict__ counts,
                                                   const int* __restrict__ tok_list,
                                                   const float* __restrict__ gate_list,
                                                   float* __restrict__ out) {
    int e = blockIdx.z;
    int cnt = counts[e];
    int mt = blockIdx.y;
    if (mt * 16 >= cnt) return;
    int n0 = blockIdx.x * 64;
    int tid = threadIdx.x, lane = tid & 31, wave = tid >> 5;

    __shared__ __align__(16) __bf16 As[16 * 32];
    __shared__ __align__(16) __bf16 Bt[64 * 32];

    v8f acc = {0, 0, 0, 0, 0, 0, 0, 0};

    int arow = tid >> 3, acol = (tid & 7) * 4;
    const __bf16* atile0 = act + ((size_t)e * T_NUM + mt * 16) * F_DIM;   // tile origin
    const __bf16* actrow = atile0 + (size_t)arow * F_DIM + acol;          // fallback
    const float* w2base = w2 + (size_t)e * F_DIM * H_DIM + n0;

#pragma unroll 2
    for (int k0 = 0; k0 < F_DIM; k0 += 32) {
#ifdef HAVE_TDM
        if (tid == 0)
            tdm_load_2d_bf16(atile0 + k0, As, 32, 16, F_DIM);
        (void)actrow;
#else
        *(uint2*)&As[arow * 32 + acol] = *(const uint2*)(actrow + k0);
#endif
#pragma unroll
        for (int i = 0; i < 2; ++i) {
            int unit = tid + i * 128;
            int rp = unit >> 4;
            int c4 = (unit & 15) * 4;
            const float* r0 = w2base + (size_t)(k0 + 2 * rp) * H_DIM + c4;
            float4 a0 = *(const float4*)r0;
            float4 a1 = *(const float4*)(r0 + H_DIM);
            *(bf16x2*)&Bt[(c4 + 0) * 32 + 2 * rp] = bf16x2{(__bf16)a0.x, (__bf16)a1.x};
            *(bf16x2*)&Bt[(c4 + 1) * 32 + 2 * rp] = bf16x2{(__bf16)a0.y, (__bf16)a1.y};
            *(bf16x2*)&Bt[(c4 + 2) * 32 + 2 * rp] = bf16x2{(__bf16)a0.z, (__bf16)a1.z};
            *(bf16x2*)&Bt[(c4 + 3) * 32 + 2 * rp] = bf16x2{(__bf16)a0.w, (__bf16)a1.w};
        }
        if (k0 + 32 < F_DIM)
            __builtin_prefetch(w2base + (size_t)(k0 + 32) * H_DIM, 0, 1);
        __syncthreads();
        v16bf a = load_a_frag(As, lane);
        v16bf b = load_bt_frag(Bt, wave * 16, lane);
        acc = __builtin_amdgcn_wmma_f32_16x16x32_bf16(false, a, false, b, (short)0, acc, false, false);
        __syncthreads();
    }

    int half = lane >> 4, n = lane & 15;
#pragma unroll
    for (int j = 0; j < 8; ++j) {
        int m = j + 8 * half;
        int g = mt * 16 + m;
        if (g < cnt) {
            int tok = tok_list[e * T_NUM + g];
            float gate = gate_list[e * T_NUM + g];
            atomicAdd(&out[(size_t)tok * H_DIM + n0 + wave * 16 + n], gate * acc[j]);
        }
    }
}

// ---------------------------------------------------------------------------
extern "C" void kernel_launch(void* const* d_in, const int* in_sizes, int n_in,
                              void* d_out, int out_size, void* d_ws, size_t ws_size,
                              hipStream_t stream) {
    (void)in_sizes; (void)n_in; (void)out_size; (void)ws_size;

    const float* x  = (const float*)d_in[0];  // [T, H]
    const float* rw = (const float*)d_in[1];  // [E, H]
    const float* w1 = (const float*)d_in[2];  // [E, H, F]
    const float* v1 = (const float*)d_in[3];  // [E, H, F]
    const float* w2 = (const float*)d_in[4];  // [E, F, H]
    const float* sg = (const float*)d_in[5];  // [FS, H]
    const float* su = (const float*)d_in[6];  // [FS, H]
    const float* sd = (const float*)d_in[7];  // [H, FS]
    float* out = (float*)d_out;               // [T, H]

    // Workspace layout
    char* ws = (char*)d_ws;
    int*    counts    = (int*)ws;                                   // 16 ints (pad 256)
    int*    tok_list  = (int*)(ws + 256);                           // E*T ints
    float*  gate_list = (float*)(ws + 256 + (size_t)E_NUM * T_NUM * 4);
    __bf16* act       = (__bf16*)(ws + 256 + 2 * (size_t)E_NUM * T_NUM * 4);       // E*T*F bf16
    __bf16* acts      = (__bf16*)((char*)act + (size_t)E_NUM * T_NUM * F_DIM * 2); // T*FS bf16

    hipMemsetAsync(counts, 0, 256, stream);

    k_router<<<T_NUM / 16, 256, 0, stream>>>(x, rw, counts, tok_list, gate_list);
    k_shared_gemm1<<<dim3(FS_DIM / 64, T_NUM / 16), 128, 0, stream>>>(x, sg, su, acts);
    k_shared_gemm2<<<dim3(H_DIM / 64, T_NUM / 16), 128, 0, stream>>>(acts, sd, out);
    k_moe_gemm1<<<dim3(F_DIM / 64, T_NUM / 16, E_NUM), 128, 0, stream>>>(x, w1, v1, counts, tok_list, act);
    k_moe_gemm2<<<dim3(H_DIM / 64, T_NUM / 16, E_NUM), 128, 0, stream>>>(act, w2, counts, tok_list, gate_list, out);
}
